// EquiformerV2Potential_32229434589411
// MI455X (gfx1250) — matrix operand order repaired and checked by reading, not compiled, not run on hardware
//
#include <hip/hip_runtime.h>
#include <hip/hip_bf16.h>

// ---------------------------------------------------------------------------
// EquiformerV2-style potential, forward pass on gfx1250 (MI455X).
// All GEMMs run on v_wmma_f32_16x16x32_bf16.
//  - A-fragments load directly from row-major global memory as 2x b128
//    (ISA layout: lane=m%16, element order k = hi..hi+7, 16+hi..16+hi+7).
//  - Weight B-fragments are pre-swizzled into fragment-tile order so each
//    lane loads 16B at +0B and 16B at +32B (2x b128), no LDS.
//  - Only AV's V-transpose and the fused bias-MLP C->A relayout use LDS,
//    with 2x 16B ds_load reads 32B apart.
// ---------------------------------------------------------------------------

#define BB   4
#define NN   384
#define HH   256
#define NHH  8
#define HDD  32
#define DDD  32
#define LL   4
#define BN   (BB * NN)
#define CUT  5.0f
#define PI_F 3.14159265358979323846f
#define FNEG (-3.402823466e38f)

typedef __bf16 bf16x16 __attribute__((ext_vector_type(16)));
typedef float  f32x8   __attribute__((ext_vector_type(8)));
typedef unsigned short u16;
typedef unsigned int   u32;

struct B32x2 { uint4 lo, hi; };

// ---- helpers ---------------------------------------------------------------

__device__ __forceinline__ u16 f2bf(float f) {
  u32 u = __float_as_uint(f);
  u32 r = u + 0x7FFFu + ((u >> 16) & 1u);  // round-to-nearest-even
  return (u16)(r >> 16);
}

// A-fragment from a row-major 32-wide K slice. p must already point at
// row (m0 + lane%16), k offset (k0 + 8*(lane>=16)).
__device__ __forceinline__ bf16x16 ld_a32(const u16* p) {
  B32x2 t;
  t.lo = *(const uint4*)p;         // k = hi .. hi+7
  t.hi = *(const uint4*)(p + 16);  // k = 16+hi .. 16+hi+7
  return __builtin_bit_cast(bf16x16, t);
}

// B-fragment from 32B-contiguous storage (pre-swizzled weights / LDS tile).
__device__ __forceinline__ bf16x16 ld_b32c(const u16* p) {
  B32x2 t;
  t.lo = *(const uint4*)p;
  t.hi = *(const uint4*)(p + 16);
  return __builtin_bit_cast(bf16x16, t);
}

__device__ __forceinline__ float act_apply(float x, int act) {
  if (act == 1) return x / (1.0f + __expf(-x));                      // silu
  if (act == 2) return 0.5f * x * (1.0f + erff(x * 0.70710678118f)); // gelu exact
  return x;
}

// ---- weight pre-swizzle ----------------------------------------------------
// W f32 [nmat][K,N] row-major -> bf16 fragment tiles [nmat][nt][kt][lane][16]:
//   k = kt*32 + 16*(e>=8) + 8*(lane>=16) + e%8 , n = nt*16 + lane%16
__global__ void k_castw(const float* __restrict__ W, u16* __restrict__ Wsz,
                        int K, int N, int nmat) {
  const int idx = blockIdx.x * 256 + threadIdx.x;
  const int per = K * N;
  if (idx >= per * nmat) return;
  const int m = idx / per;
  const int r = idx - m * per;
  const int e    = r & 15;
  const int lane = (r >> 4) & 31;
  const int tile = r >> 9;
  const int ktiles = K >> 5;
  const int kt = tile % ktiles;
  const int nt = tile / ktiles;
  const int k = kt * 32 + ((e >= 8) ? 16 : 0) + ((lane >= 16) ? 8 : 0) + (e & 7);
  const int n = nt * 16 + (lane & 15);
  Wsz[idx] = f2bf(W[(size_t)m * per + (size_t)k * N + n]);
}

// rb_w2 [L,256,8] -> swizzled padded [L][kt=0..7][lane][16] (n>=8 zero)
__global__ void k_padw2s(const float* __restrict__ rb_w2, u16* __restrict__ w2ps) {
  const int idx = blockIdx.x * 256 + threadIdx.x;
  if (idx >= LL * HH * 16) return;
  const int per = HH * 16;
  const int l = idx / per;
  const int r = idx - l * per;
  const int e    = r & 15;
  const int lane = (r >> 4) & 31;
  const int kt   = r >> 9;  // 0..7
  const int k = kt * 32 + ((e >= 8) ? 16 : 0) + ((lane >= 16) ? 8 : 0) + (e & 7);
  const int n = lane & 15;
  w2ps[idx] = (n < NHH) ? f2bf(rb_w2[((size_t)l * HH + k) * NHH + n]) : (u16)0;
}

// ---- elementwise / setup kernels ------------------------------------------

__global__ void k_embed(const int* __restrict__ idx, const float* __restrict__ mask,
                        const float* __restrict__ emb, float* __restrict__ x) {
  const int r = blockIdx.x, h = threadIdx.x;
  const float m = (mask[r] > 0.f) ? 1.f : 0.f;
  x[(size_t)r * HH + h] = emb[(size_t)idx[r] * HH + h] * m;
}

__global__ __launch_bounds__(128)
void k_geom(const float* __restrict__ pos, const float* __restrict__ mask,
            unsigned char* __restrict__ edge, u16* __restrict__ feat16) {
  __shared__ float sDist[NN];
  __shared__ int   sAny[128];
  const int bi = blockIdx.x, b = bi / NN, i = bi % NN;
  const int tid = threadIdx.x;
  const float mi = (mask[b * NN + i] > 0.f) ? 1.f : 0.f;
  const float pix = pos[(size_t)(b * NN + i) * 3 + 0] * mi;
  const float piy = pos[(size_t)(b * NN + i) * 3 + 1] * mi;
  const float piz = pos[(size_t)(b * NN + i) * 3 + 2] * mi;
  int any = 0;
  for (int j = tid; j < NN; j += 128) {
    const float mj = (mask[b * NN + j] > 0.f) ? 1.f : 0.f;
    const float dx = pix - pos[(size_t)(b * NN + j) * 3 + 0] * mj + 1e-9f;
    const float dy = piy - pos[(size_t)(b * NN + j) * 3 + 1] * mj + 1e-9f;
    const float dz = piz - pos[(size_t)(b * NN + j) * 3 + 2] * mj + 1e-9f;
    const float d = sqrtf(dx * dx + dy * dy + dz * dz);
    sDist[j] = d;
    if (d <= CUT && mi > 0.f && mj > 0.f) any = 1;
  }
  sAny[tid] = any;
  __syncthreads();
  for (int s = 64; s > 0; s >>= 1) { if (tid < s) sAny[tid] |= sAny[tid + s]; __syncthreads(); }
  const int isolated = (mi > 0.f) && (sAny[0] == 0);
  for (int j = tid; j < NN; j += 128) {
    const float mj = (mask[b * NN + j] > 0.f) ? 1.f : 0.f;
    const float d = sDist[j];
    float adjv = (d <= CUT && mi > 0.f && mj > 0.f) ? 1.f : 0.f;
    if (isolated && j == i) adjv += 1.f;
    const size_t eo = ((size_t)(b * NN + i)) * NN + j;
    edge[eo] = (adjv > 0.f) ? 1 : 0;
    u16* fp = &feat16[eo * DDD];
    const float fcut = (d <= CUT) ? 1.f : 0.f;
    const float inv  = 1.f / (d + 1e-6f);
    const float wph  = PI_F * d / CUT;
    const float sc   = inv * fcut * adjv;
#pragma unroll 4
    for (int k = 0; k < DDD; ++k)
      fp[k] = f2bf(sinf(wph * (float)(k + 1)) * sc);
  }
}

__global__ __launch_bounds__(256)
void k_ln(const float* __restrict__ xin, const float* __restrict__ g,
          const float* __restrict__ bta, u16* __restrict__ out16) {
  __shared__ float red[256];
  const int r = blockIdx.x, h = threadIdx.x;
  const float v = xin[(size_t)r * HH + h];
  red[h] = v; __syncthreads();
  for (int s = 128; s > 0; s >>= 1) { if (h < s) red[h] += red[h + s]; __syncthreads(); }
  const float mean = red[0] / (float)HH; __syncthreads();
  const float c = v - mean;
  red[h] = c * c; __syncthreads();
  for (int s = 128; s > 0; s >>= 1) { if (h < s) red[h] += red[h + s]; __syncthreads(); }
  const float var = red[0] / (float)HH;
  out16[(size_t)r * HH + h] = f2bf(c * rsqrtf(var + 1e-5f) * g[h] + bta[h]);
}

__global__ void k_ctxgate(const float* __restrict__ ctx, const float* __restrict__ gated,
                          u16* __restrict__ out16, int n) {
  int i = blockIdx.x * 256 + threadIdx.x;
  if (i < n) {
    const float s = 1.f / (1.f + __expf(-gated[i]));
    out16[i] = f2bf(ctx[i] * s);
  }
}

// ---- generic WMMA GEMM: out = act(A[M,K]@W[K,N] + bias) (+res) -------------
// A bf16 [M,lda] row-major, Wsz pre-swizzled fragment tiles.
// 8 waves/block, one 16x16 tile/wave, no LDS, no barriers.
__global__ __launch_bounds__(256)
void k_gemm(const u16* __restrict__ A, int lda,
            const u16* __restrict__ Wsz, int ktiles,
            const float* __restrict__ bias,
            const float* __restrict__ res, int ldres,
            float* __restrict__ outF, u16* __restrict__ out16, int ldc,
            int act) {
  const int tid = threadIdx.x, lane = tid & 31, w = tid >> 5;
  const int m0 = blockIdx.y * 16;
  const int nt = blockIdx.x * 8 + w;
  const int n0 = nt * 16;
  const u16* wt = Wsz + (size_t)nt * ktiles * 512 + lane * 16;
  const u16* ar = A + (size_t)(m0 + (lane & 15)) * lda + ((lane >= 16) ? 8 : 0);
  f32x8 acc = {0, 0, 0, 0, 0, 0, 0, 0};
  for (int kt = 0; kt < ktiles; ++kt) {
    const bf16x16 a = ld_a32(ar);
    const bf16x16 b = ld_b32c(wt);
    if (kt + 1 < ktiles) {  // gfx1250 global_prefetch_b8 of next K tile
      __builtin_prefetch(wt + 512, 0, 3);
      __builtin_prefetch(ar + 32, 0, 3);
    }
    acc = __builtin_amdgcn_wmma_f32_16x16x32_bf16(false, a, false, b, (short)0, acc, false, false);
    ar += 32;
    wt += 512;
  }
#pragma unroll
  for (int r = 0; r < 8; ++r) {
    const int m = m0 + r + ((lane >= 16) ? 8 : 0);
    const int n = n0 + (lane & 15);
    float v = acc[r];
    if (bias) v += bias[n];
    v = act_apply(v, act);
    if (res) v += res[(size_t)m * ldres + n];
    if (outF)  outF[(size_t)m * ldc + n] = v;
    if (out16) out16[(size_t)m * ldc + n] = f2bf(v);
  }
}

// ---- fused rb bias MLP: logits[b,h,i,j] = silu(feat@w1+b1)@w2 + b2 ---------
// One wave per (b,i,jtile): 16 WMMAs (GEMM1 over H) + 8 WMMAs (GEMM2, K=256).
// w1sz: 16 fragment tiles [nt][512]; w2ps: 8 fragment tiles [kt][512].
__global__ __launch_bounds__(32)
void k_bias(const u16* __restrict__ feat16,
            const u16* __restrict__ w1sz,
            const float* __restrict__ b1,
            const u16* __restrict__ w2ps,
            const float* __restrict__ b2,
            float* __restrict__ logits) {
  __shared__ alignas(16) u16 sH[512];
  const int lane = threadIdx.x;
  const int t  = blockIdx.x;
  const int jt = t % (NN / 16);
  const int i  = (t / (NN / 16)) % NN;
  const int b  = t / ((NN / 16) * NN);
  const int j0 = jt * 16;

  // A1 direct from feat rows (row-major, 32 contiguous k)
  const u16* fr = feat16 + (((size_t)(b * NN + i)) * NN + j0 + (lane & 15)) * DDD
                + ((lane >= 16) ? 8 : 0);
  const bf16x16 a1 = ld_a32(fr);

  f32x8 acc2 = {0, 0, 0, 0, 0, 0, 0, 0};
  const int n  = lane & 15;
  const int hi = (n >= 8) ? 1 : 0;
  const int kp = n - hi * 8;
  for (int c = 0; c < 8; ++c) {
#pragma unroll
    for (int sub = 0; sub < 2; ++sub) {
      const int ntile = 2 * c + sub;
      const bf16x16 bw = ld_b32c(w1sz + (size_t)ntile * 512 + lane * 16);
      f32x8 c1 = {0, 0, 0, 0, 0, 0, 0, 0};
      c1 = __builtin_amdgcn_wmma_f32_16x16x32_bf16(false, a1, false, bw, (short)0, c1, false, false);
      const float bb = b1[ntile * 16 + n];
      const int e = 2 * (sub * 4 + (kp >> 1)) + (kp & 1);  // C->A relayout target
#pragma unroll
      for (int r = 0; r < 8; ++r) {
        const int mrow = r + ((lane >= 16) ? 8 : 0);
        float vv = c1[r] + bb;
        vv = vv / (1.0f + __expf(-vv));  // silu
        sH[(mrow + hi * 16) * 16 + e] = f2bf(vv);
      }
    }
    __syncthreads();
    const bf16x16 a2  = ld_b32c(&sH[lane * 16]);  // 2x 16B LDS reads at +0B and +32B
    const bf16x16 b2f = ld_b32c(w2ps + (size_t)c * 512 + lane * 16);
    acc2 = __builtin_amdgcn_wmma_f32_16x16x32_bf16(false, a2, false, b2f, (short)0, acc2, false, false);
    __syncthreads();
  }
#pragma unroll
  for (int r = 0; r < 8; ++r) {
    const int j = j0 + r + ((lane >= 16) ? 8 : 0);
    if (n < NHH) {
      const size_t o = (((size_t)(b * NHH + n)) * NN + i) * NN + j;
      logits[o] = acc2[r] + b2[n];
    }
  }
}

// ---- QK^T: logits += q.k / sqrt(HD). No LDS: both fragments direct ---------
__global__ __launch_bounds__(256)
void k_qk(const u16* __restrict__ qkv16, float* __restrict__ logits, float invscale) {
  const int tid = threadIdx.x, lane = tid & 31, w = tid >> 5;
  const int bh = blockIdx.x, b = bh / NHH, h = bh % NHH;
  const int i0 = blockIdx.y * 16;
  const int j0 = (blockIdx.z * 8 + w) * 16;
  // A = q tile rows i
  const u16* qr = qkv16 + ((size_t)(b * NN + i0 + (lane & 15))) * 768 + h * 32
                + ((lane >= 16) ? 8 : 0);
  const bf16x16 a = ld_a32(qr);
  // B = k^T tile: lane holds column j=lane%16, elements d = 8*(lane>=16)+e%8 (+16 for e>=8)
  // -> two 8-wide runs of d values of k-row j. No transpose needed.
  const u16* kr = qkv16 + ((size_t)(b * NN + j0 + (lane & 15))) * 768 + 256 + h * 32
                + ((lane >= 16) ? 8 : 0);
  const bf16x16 bt = ld_b32c(kr);
  f32x8 acc = {0, 0, 0, 0, 0, 0, 0, 0};
  acc = __builtin_amdgcn_wmma_f32_16x16x32_bf16(false, a, false, bt, (short)0, acc, false, false);
#pragma unroll
  for (int r = 0; r < 8; ++r) {
    const int i = i0 + r + ((lane >= 16) ? 8 : 0);
    const int j = j0 + (lane & 15);
    const size_t o = (((size_t)(b * NHH + h)) * NN + i) * NN + j;
    logits[o] += acc[r] * invscale;
  }
}

// ---- masked softmax over j, output bf16 attn -------------------------------
__global__ __launch_bounds__(128)
void k_softmax(const float* __restrict__ logits, const unsigned char* __restrict__ edge,
               u16* __restrict__ attn16) {
  __shared__ float red[128];
  const int row = blockIdx.x;  // (b*NH + h)*NN + i
  const int i = row % NN;
  const int bh = row / NN, b = bh / NHH;
  const int tid = threadIdx.x;
  const float* lp = &logits[(size_t)row * NN];
  const unsigned char* ep = &edge[((size_t)(b * NN + i)) * NN];
  float mx = FNEG;
  for (int j = tid; j < NN; j += 128) { const float v = ep[j] ? lp[j] : FNEG; mx = fmaxf(mx, v); }
  red[tid] = mx; __syncthreads();
  for (int s = 64; s > 0; s >>= 1) { if (tid < s) red[tid] = fmaxf(red[tid], red[tid + s]); __syncthreads(); }
  mx = red[0]; __syncthreads();
  float sum = 0.f;
  for (int j = tid; j < NN; j += 128) { const float v = ep[j] ? lp[j] : FNEG; sum += __expf(v - mx); }
  red[tid] = sum; __syncthreads();
  for (int s = 64; s > 0; s >>= 1) { if (tid < s) red[tid] += red[tid + s]; __syncthreads(); }
  const float inv = 1.f / red[0];
  u16* ap = &attn16[(size_t)row * NN];
  for (int j = tid; j < NN; j += 128) {
    const float v = ep[j] ? lp[j] : FNEG;
    ap[j] = f2bf(__expf(v - mx) * inv);
  }
}

// ---- AV: ctx[b,i,h*32+d] = sum_j attn * v ---------------------------------
// A (attn rows) direct; V needs a true transpose -> LDS, contiguous reads.
__global__ __launch_bounds__(64)
void k_av(const u16* __restrict__ attn16, const u16* __restrict__ qkv16,
          float* __restrict__ ctx) {
  __shared__ alignas(16) u16 sB[2][512];
  const int tid = threadIdx.x, lane = tid & 31, w = tid >> 5;
  const int bh = blockIdx.x, b = bh / NHH, h = bh % NHH;
  const int i0 = blockIdx.y * 16;
  const int d0 = w * 16;
  const u16* ar = attn16 + ((size_t)bh * NN + i0 + (lane & 15)) * NN + ((lane >= 16) ? 8 : 0);
  const int hi16 = (lane & 8) ? 16 : 0;
  const int e = (lane & 7) + ((lane >= 16) ? 8 : 0);
  f32x8 acc = {0, 0, 0, 0, 0, 0, 0, 0};
  for (int k0 = 0; k0 < NN; k0 += 32) {
    {  // transpose-stage V rows: staging lane holds row k=j_local=lane
      const u16* src = qkv16 + ((size_t)(b * NN + k0 + lane)) * 768 + 512 + h * 32 + d0;
#pragma unroll
      for (int nn = 0; nn < 16; ++nn) sB[w][(nn + hi16) * 16 + e] = src[nn];
    }
    __syncthreads();
    const bf16x16 a = ld_a32(ar + k0);
    const bf16x16 bt = ld_b32c(&sB[w][lane * 16]);
    acc = __builtin_amdgcn_wmma_f32_16x16x32_bf16(false, a, false, bt, (short)0, acc, false, false);
    __syncthreads();
  }
#pragma unroll
  for (int r = 0; r < 8; ++r) {
    const int i = i0 + r + ((lane >= 16) ? 8 : 0);
    ctx[((size_t)(b * NN + i)) * HH + h * 32 + d0 + (lane & 15)] = acc[r];
  }
}

// ---- pooling + energy head -------------------------------------------------
__global__ __launch_bounds__(256)
void k_pool(const float* __restrict__ pooled, const float* __restrict__ mask,
            const float* __restrict__ ehw, const float* __restrict__ ehb,
            float* __restrict__ out) {
  __shared__ float red[256];
  const int b = blockIdx.x, h = threadIdx.x;
  float cnt = 0.f;
  for (int n2 = h; n2 < NN; n2 += 256) cnt += (mask[b * NN + n2] > 0.f) ? 1.f : 0.f;
  red[h] = cnt; __syncthreads();
  for (int s = 128; s > 0; s >>= 1) { if (h < s) red[h] += red[h + s]; __syncthreads(); }
  const float count = fmaxf(red[0], 1.f); __syncthreads();
  float s = 0.f;
  for (int n2 = 0; n2 < NN; ++n2) {
    const float m = (mask[b * NN + n2] > 0.f) ? 1.f : 0.f;
    s += pooled[((size_t)(b * NN + n2)) * HH + h] * m;
  }
  red[h] = (s / count) * ehw[h]; __syncthreads();
  for (int sq = 128; sq > 0; sq >>= 1) { if (h < sq) red[h] += red[h + sq]; __syncthreads(); }
  if (h == 0) out[b] = red[0] + ehb[0];
}

// ---------------------------------------------------------------------------

extern "C" void kernel_launch(void* const* d_in, const int* in_sizes, int n_in,
                              void* d_out, int out_size, void* d_ws, size_t ws_size,
                              hipStream_t stream) {
  (void)in_sizes; (void)n_in; (void)out_size; (void)ws_size;
  const int*   node_idx  = (const int*)  d_in[0];
  const float* positions = (const float*)d_in[1];
  const float* mask      = (const float*)d_in[2];
  const float* emb       = (const float*)d_in[3];
  const float* ln1_g = (const float*)d_in[4],  *ln1_b = (const float*)d_in[5];
  const float* qkv_w = (const float*)d_in[6],  *qkv_b = (const float*)d_in[7];
  const float* out_w = (const float*)d_in[8],  *out_b = (const float*)d_in[9];
  const float* rb_w1 = (const float*)d_in[10], *rb_b1 = (const float*)d_in[11];
  const float* rb_w2 = (const float*)d_in[12], *rb_b2 = (const float*)d_in[13];
  const float* gate_w1 = (const float*)d_in[14], *gate_b1 = (const float*)d_in[15];
  const float* gate_w2 = (const float*)d_in[16], *gate_b2 = (const float*)d_in[17];
  const float* ln2_g = (const float*)d_in[18], *ln2_b = (const float*)d_in[19];
  const float* ff_w1 = (const float*)d_in[20], *ff_b1 = (const float*)d_in[21];
  const float* ff_w2 = (const float*)d_in[22], *ff_b2 = (const float*)d_in[23];
  const float* pool_g = (const float*)d_in[24], *pool_beta = (const float*)d_in[25];
  const float* pool_w = (const float*)d_in[26], *pool_b = (const float*)d_in[27];
  const float* eh_w = (const float*)d_in[28], *eh_b = (const float*)d_in[29];
  float* out = (float*)d_out;

  char* ws = (char*)d_ws;
  size_t off = 0;
  auto alloc = [&](size_t bytes) -> char* {
    char* p = ws + off; off = (off + bytes + 255) & ~(size_t)255; return p;
  };
  float* xF      = (float*)alloc((size_t)BN * HH * 4);
  u16*   h16     = (u16*)  alloc((size_t)BN * HH * 2);
  u16*   qkv16   = (u16*)  alloc((size_t)BN * 768 * 2);
  u16*   g116    = (u16*)  alloc((size_t)BN * HH * 2);
  u16*   f116    = (u16*)  alloc((size_t)BN * 512 * 2);
  float* gatedF  = (float*)alloc((size_t)BN * HH * 4);
  float* ctxF    = (float*)alloc((size_t)BN * HH * 4);
  u16*   ctxg16  = (u16*)  alloc((size_t)BN * HH * 2);
  u16*   feat16  = (u16*)  alloc((size_t)BB * NN * NN * DDD * 2);
  unsigned char* edgeB = (unsigned char*)alloc((size_t)BB * NN * NN);
  float* logitsF = (float*)alloc((size_t)BB * NHH * NN * NN * 4);
  u16*   attn16  = (u16*)  alloc((size_t)BB * NHH * NN * NN * 2);
  float* pooledF = (float*)alloc((size_t)BN * HH * 4);
  u16* qkv_w16   = (u16*)alloc((size_t)LL * HH * 768 * 2);
  u16* out_w16   = (u16*)alloc((size_t)LL * HH * HH * 2);
  u16* rb_w1_16  = (u16*)alloc((size_t)LL * DDD * HH * 2);
  u16* rb_w2p16  = (u16*)alloc((size_t)LL * HH * 16 * 2);
  u16* gate_w1_16 = (u16*)alloc((size_t)LL * HH * HH * 2);
  u16* gate_w2_16 = (u16*)alloc((size_t)LL * HH * HH * 2);
  u16* ff_w1_16  = (u16*)alloc((size_t)LL * HH * 512 * 2);
  u16* ff_w2_16  = (u16*)alloc((size_t)LL * 512 * HH * 2);
  u16* pool_w16  = (u16*)alloc((size_t)HH * HH * 2);

  auto castw = [&](const float* src, u16* dst, int K, int N, int nmat) {
    size_t tot = (size_t)K * N * nmat;
    k_castw<<<(unsigned)((tot + 255) / 256), 256, 0, stream>>>(src, dst, K, N, nmat);
  };
  castw(qkv_w,   qkv_w16,   HH, 768, LL);
  castw(out_w,   out_w16,   HH, HH,  LL);
  castw(rb_w1,   rb_w1_16,  DDD, HH, LL);
  castw(gate_w1, gate_w1_16, HH, HH, LL);
  castw(gate_w2, gate_w2_16, HH, HH, LL);
  castw(ff_w1,   ff_w1_16,  HH, 512, LL);
  castw(ff_w2,   ff_w2_16,  512, HH, LL);
  castw(pool_w,  pool_w16,  HH, HH, 1);
  k_padw2s<<<(LL * HH * 16 + 255) / 256, 256, 0, stream>>>(rb_w2, rb_w2p16);

  k_embed<<<BN, HH, 0, stream>>>(node_idx, mask, emb, xF);
  k_geom<<<BB * NN, 128, 0, stream>>>(positions, mask, edgeB, feat16);

  const float invscale = 0.17677669529663687f;  // 1/sqrt(32)
  for (int l = 0; l < LL; ++l) {
    k_ln<<<BN, 256, 0, stream>>>(xF, ln1_g + l * HH, ln1_b + l * HH, h16);
    // qkv: [1536,256]@[256,768]
    k_gemm<<<dim3(768 / 128, BN / 16), 256, 0, stream>>>(
        h16, HH, qkv_w16 + (size_t)l * HH * 768, HH / 32,
        qkv_b + l * 768, nullptr, 0, nullptr, qkv16, 768, 0);
    // fused rb bias MLP -> logits
    k_bias<<<BB * NN * (NN / 16), 32, 0, stream>>>(
        feat16, rb_w1_16 + (size_t)l * DDD * HH, rb_b1 + l * HH,
        rb_w2p16 + (size_t)l * HH * 16, rb_b2 + l * NHH, logitsF);
    // q.k^T / scale (accumulates into logits)
    k_qk<<<dim3(BB * NHH, NN / 16, (NN / 16) / 8), 256, 0, stream>>>(qkv16, logitsF, invscale);
    k_softmax<<<BB * NHH * NN, 128, 0, stream>>>(logitsF, edgeB, attn16);
    k_av<<<dim3(BB * NHH, NN / 16), 64, 0, stream>>>(attn16, qkv16, ctxF);
    // gate MLP
    k_gemm<<<dim3(HH / 128, BN / 16), 256, 0, stream>>>(
        h16, HH, gate_w1_16 + (size_t)l * HH * HH, HH / 32,
        gate_b1 + l * HH, nullptr, 0, nullptr, g116, HH, 1);
    k_gemm<<<dim3(HH / 128, BN / 16), 256, 0, stream>>>(
        g116, HH, gate_w2_16 + (size_t)l * HH * HH, HH / 32,
        gate_b2 + l * HH, nullptr, 0, gatedF, nullptr, HH, 0);
    k_ctxgate<<<(BN * HH + 255) / 256, 256, 0, stream>>>(ctxF, gatedF, ctxg16, BN * HH);
    // out projection + residual (in-place: each element read/written by one lane)
    k_gemm<<<dim3(HH / 128, BN / 16), 256, 0, stream>>>(
        ctxg16, HH, out_w16 + (size_t)l * HH * HH, HH / 32,
        out_b + l * HH, xF, HH, xF, nullptr, HH, 0);
    // FFN
    k_ln<<<BN, 256, 0, stream>>>(xF, ln2_g + l * HH, ln2_b + l * HH, h16);
    k_gemm<<<dim3(512 / 128, BN / 16), 256, 0, stream>>>(
        h16, HH, ff_w1_16 + (size_t)l * HH * 512, HH / 32,
        ff_b1 + l * 512, nullptr, 0, nullptr, f116, 512, 2);
    k_gemm<<<dim3(HH / 128, BN / 16), 256, 0, stream>>>(
        f116, 512, ff_w2_16 + (size_t)l * 512 * HH, 512 / 32,
        ff_b2 + l * HH, xF, HH, xF, nullptr, HH, 0);
  }

  k_ln<<<BN, 256, 0, stream>>>(xF, pool_g, pool_beta, h16);
  k_gemm<<<dim3(HH / 128, BN / 16), 256, 0, stream>>>(
      h16, HH, pool_w16, HH / 32, pool_b, nullptr, 0, pooledF, nullptr, HH, 1);
  k_pool<<<BB, 256, 0, stream>>>(pooledF, mask, eh_w, eh_b, out);
}
